// MiNer2_73976516706887
// MI455X (gfx1250) — matrix operand order repaired and checked by reading, not compiled, not run on hardware
//
#include <hip/hip_runtime.h>

typedef __attribute__((ext_vector_type(16))) _Float16 v16h;
typedef __attribute__((ext_vector_type(8)))  _Float16 v8h;
typedef __attribute__((ext_vector_type(8)))  float    v8f;

#define NUM_RELS  500
#define HIDDEN    200
#define NUM_TYPES 50
#define HEADS     5
#define N1        20000
#define N2        20000
#define D0        20
#define D1        20
#define L_MB      21      // mailbox rows (20 msgs + 1 mean)
#define KP        224     // K padded to 7 * 32 (logical)
#define AST       232     // A-tile LDS row stride in f16 (116 dwords: bank-conflict-free)
#define NP        64      // N padded to 4 * 16
#define MP        32      // M padded to 2 * 16

// ---------------------------------------------------------------------------
// Kernel 0: B = fc_w^T as f16, padded [KP][NP]
// ---------------------------------------------------------------------------
__global__ __launch_bounds__(256) void prep_B_kernel(const float* __restrict__ fc_w,
                                                     _Float16* __restrict__ B) {
    int i = blockIdx.x * 256 + threadIdx.x;
    if (i >= KP * NP) return;
    int k = i / NP, n = i % NP;
    float v = (k < HIDDEN && n < NUM_TYPES) ? fc_w[n * HIDDEN + k] : 0.0f;
    B[i] = (_Float16)v;
}

// ---------------------------------------------------------------------------
// Kernel 1: agg[n, :] = mean_e ( entity[ids0[src]] + sign * relation[r] )
// One block per dst node; thread j owns hidden column j (j < 200).
// ---------------------------------------------------------------------------
__global__ __launch_bounds__(256) void agg_kernel(const float* __restrict__ entity,
                                                  const float* __restrict__ relation,
                                                  const int* __restrict__ ids0,
                                                  const int* __restrict__ es0,
                                                  const int* __restrict__ et0,
                                                  float* __restrict__ agg) {
    int n = blockIdx.x;
    int j = threadIdx.x;
    if (j >= HIDDEN) return;
    float acc = 0.0f;
    for (int e = 0; e < D0; ++e) {
        int idx = n * D0 + e;
        int s   = es0[idx];
        int ty  = et0[idx];
        float sg = (ty >= NUM_RELS) ? -1.0f : 1.0f;
        int r   = ty % NUM_RELS;
        int eid = ids0[s];
        acc += entity[(size_t)eid * HIDDEN + j] + sg * relation[r * HIDDEN + j];
    }
    agg[(size_t)n * HIDDEN + j] = acc * (1.0f / D0);
}

// ---------------------------------------------------------------------------
// Kernel 2: fused mailbox build + WMMA fc + MHA + blend + sigmoid.
// One block (128 threads / 4 waves) per dst node.
// ---------------------------------------------------------------------------
__global__ __launch_bounds__(128) void mti_kernel(const float* __restrict__ entity,
                                                  const float* __restrict__ relation,
                                                  const float* __restrict__ fc_b,
                                                  const int* __restrict__ ids1,
                                                  const int* __restrict__ es1,
                                                  const int* __restrict__ et1,
                                                  const float* __restrict__ agg,
                                                  const _Float16* __restrict__ Bg,
                                                  float* __restrict__ out) {
    __shared__ _Float16 Amsg[MP * AST];  // relu(mailbox msg) f16, K-padded, stride 232
    __shared__ _Float16 Aagg[MP * AST];  // relu(mailbox agg) f16
    __shared__ float    P1[MP * NP];     // fc(msg rows) f32
    __shared__ float    P2[MP * NP];     // fc(aggm rows) f32
    __shared__ float    qv[2][NP];       // per-head mean query, flattened d
    __shared__ float    sc[2][HEADS][L_MB]; // scores -> attn weights

    const int n    = blockIdx.x;
    const int t    = threadIdx.x;
    const int lane = t & 31;
    const int wave = t >> 5;

    // zero both A tiles (covers K-padding columns; garbage M rows are ignored)
    {
        uint32_t* a0 = (uint32_t*)Amsg;
        uint32_t* a1 = (uint32_t*)Aagg;
        for (int i = t; i < MP * AST / 2; i += 128) { a0[i] = 0u; a1[i] = 0u; }
    }
    __syncthreads();

    // ---- mailbox build: rows 0..19, plus running raw sums for the mean row
    float sm0 = 0.f, sm1 = 0.f, sa0 = 0.f, sa1 = 0.f;
    for (int l = 0; l < D1; ++l) {
        int idx = n * D1 + l;
        int s   = es1[idx];
        int ty  = et1[idx];
        float sg = (ty >= NUM_RELS) ? -1.0f : 1.0f;
        int r   = ty % NUM_RELS;
        int eid = ids1[s];
        const float* erow = entity   + (size_t)eid * HIDDEN;
        const float* rrow = relation + (size_t)r   * HIDDEN;
        const float* grow = agg      + (size_t)s   * HIDDEN;
        {
            int j = t;                               // j in [0,128)
            float rel = sg * rrow[j];
            float vm  = erow[j] + rel;
            float va  = grow[j] + rel;
            Amsg[l * AST + j] = (_Float16)fmaxf(vm, 0.0f);
            Aagg[l * AST + j] = (_Float16)fmaxf(va, 0.0f);
            sm0 += vm; sa0 += va;
        }
        int j = t + 128;                             // j in [128,200)
        if (j < HIDDEN) {
            float rel = sg * rrow[j];
            float vm  = erow[j] + rel;
            float va  = grow[j] + rel;
            Amsg[l * AST + j] = (_Float16)fmaxf(vm, 0.0f);
            Aagg[l * AST + j] = (_Float16)fmaxf(va, 0.0f);
            sm1 += vm; sa1 += va;
        }
    }
    {   // row 20 = relu(mean of raw mailbox)
        int j = t;
        Amsg[20 * AST + j] = (_Float16)fmaxf(sm0 * (1.0f / D1), 0.0f);
        Aagg[20 * AST + j] = (_Float16)fmaxf(sa0 * (1.0f / D1), 0.0f);
        j = t + 128;
        if (j < HIDDEN) {
            Amsg[20 * AST + j] = (_Float16)fmaxf(sm1 * (1.0f / D1), 0.0f);
            Aagg[20 * AST + j] = (_Float16)fmaxf(sa1 * (1.0f / D1), 0.0f);
        }
    }
    __syncthreads();

    // ---- WMMA GEMM: P = relu(A)[32x224] * B[224x64]  (f16 in, f32 acc)
    // wave w owns N columns [16w, 16w+16). B fragments live in registers.
    const int nb = wave * 16;
    v16h bfrag[7];
#pragma unroll
    for (int kt = 0; kt < 7; ++kt) {
        // B fragment: lane L -> col nb+(L&15), K chunks {kb..kb+7, kb+16..kb+23}, kb = kt*32 + 8*(L>>4)
        const _Float16* bp = Bg + (size_t)(kt * 32 + ((lane >> 4) << 3)) * NP + nb + (lane & 15);
#pragma unroll
        for (int s = 0; s < 16; ++s)
            bfrag[kt][s] = bp[(size_t)(s < 8 ? s : s + 8) * NP];
    }
    float bias = 0.0f;
    {
        int col = nb + (lane & 15);
        if (col < NUM_TYPES) bias = fc_b[col];
    }

#pragma unroll
    for (int mat = 0; mat < 2; ++mat) {
        const _Float16* A = mat ? Aagg : Amsg;
        float*          P = mat ? P2   : P1;
#pragma unroll
        for (int mt = 0; mt < 2; ++mt) {
            const int mb = mt * 16;
            v8f acc = {};
#pragma unroll
            for (int kt = 0; kt < 7; ++kt) {
                // A fragment: lane L<16 -> row mb+L, K chunks {kb..kb+7, kb+16..kb+23};
                // lanes 16..31 same rows with +8 K offset.
                int row = mb + (lane & 15);
                int kb  = kt * 32 + ((lane >> 4) << 3);
                const _Float16* ap = A + row * AST + kb;
                v8h lo = *(const v8h*)ap;
                v8h hi = *(const v8h*)(ap + 16);
                v16h afrag = __builtin_shufflevector(lo, hi,
                                0,1,2,3,4,5,6,7,8,9,10,11,12,13,14,15);
                acc = __builtin_amdgcn_wmma_f32_16x16x32_f16(
                        false, afrag, false, bfrag[kt], (short)0, acc, false, false);
            }
            // D scatter (+bias): VGPR r -> row mb + (lane>=16?8:0) + r, col nb+(lane&15)
            int col   = nb + (lane & 15);
            int rbase = mb + ((lane >> 4) << 3);
#pragma unroll
            for (int r = 0; r < 8; ++r)
                P[(rbase + r) * NP + col] = acc[r] + bias;
        }
    }
    __syncthreads();

    // ---- parameter-free MHA over the 21 mailbox slots (heads=5, dh=10)
    const float scale = 0.31622776601683794f;  // 10^-0.5 / TEMP
    if (t < 2 * NUM_TYPES) {                   // per-column query mean
        int mat = t / NUM_TYPES, d = t % NUM_TYPES;
        const float* P = mat ? P2 : P1;
        float s = 0.0f;
        for (int l = 0; l < L_MB; ++l) s += P[l * NP + d];
        qv[mat][d] = s * (1.0f / L_MB);
    }
    __syncthreads();
    for (int i = t; i < 2 * HEADS * L_MB; i += 128) {   // scores
        int mat = i / (HEADS * L_MB);
        int rem = i % (HEADS * L_MB);
        int h = rem / L_MB, l = rem % L_MB;
        const float* P  = mat ? P2 : P1;
        const float* qq = qv[mat] + h * 10;
        const float* pr = P + l * NP + h * 10;
        float s = 0.0f;
        for (int j = 0; j < 10; ++j) s += qq[j] * pr[j];
        sc[mat][h][l] = s * scale;
    }
    __syncthreads();
    if (t < 2 * HEADS) {                        // softmax over the 21 slots
        int mat = t / HEADS, h = t % HEADS;
        float* row = sc[mat][h];
        float m = row[0];
        for (int l = 1; l < L_MB; ++l) m = fmaxf(m, row[l]);
        float ssum = 0.0f;
        for (int l = 0; l < L_MB; ++l) { float e = __expf(row[l] - m); row[l] = e; ssum += e; }
        float inv = 1.0f / ssum;
        for (int l = 0; l < L_MB; ++l) row[l] *= inv;
    }
    __syncthreads();
    if (t < NUM_TYPES) {                        // weighted sum, blend, sigmoid
        int h = t / 10;
        float o1 = 0.0f, o2 = 0.0f;
        for (int l = 0; l < L_MB; ++l) {
            o1 += sc[0][h][l] * P1[l * NP + t];
            o2 += sc[1][h][l] * P2[l * NP + t];
        }
        float pred = 0.3f * o1 + 0.7f * o2;
        out[(size_t)n * NUM_TYPES + t] = 1.0f / (1.0f + __expf(-pred));
    }
}

// ---------------------------------------------------------------------------
extern "C" void kernel_launch(void* const* d_in, const int* in_sizes, int n_in,
                              void* d_out, int out_size, void* d_ws, size_t ws_size,
                              hipStream_t stream) {
    const float* entity   = (const float*)d_in[0];
    const float* relation = (const float*)d_in[1];
    const float* fc_w     = (const float*)d_in[2];
    const float* fc_b     = (const float*)d_in[3];
    const int*   ids0     = (const int*)d_in[4];
    const int*   es0      = (const int*)d_in[5];
    const int*   et0      = (const int*)d_in[6];
    const int*   ids1     = (const int*)d_in[7];
    const int*   es1      = (const int*)d_in[8];
    const int*   et1      = (const int*)d_in[9];
    float* out = (float*)d_out;

    float*    agg = (float*)d_ws;                                  // 16 MB
    _Float16* Bg  = (_Float16*)((char*)d_ws + (size_t)N1 * HIDDEN * sizeof(float));

    prep_B_kernel<<<(KP * NP + 255) / 256, 256, 0, stream>>>(fc_w, Bg);
    agg_kernel<<<N1, 256, 0, stream>>>(entity, relation, ids0, es0, et0, agg);
    mti_kernel<<<N2, 128, 0, stream>>>(entity, relation, fc_b,
                                       ids1, es1, et1, agg, Bg, out);
}
